// RPN__29497835389412
// MI455X (gfx1250) — matrix operand order repaired and checked by reading, not compiled, not run on hardware
//
#include <hip/hip_runtime.h>
#include <hip/hip_bf16.h>
#include <stdint.h>

typedef _Float16 half_t;
typedef __attribute__((ext_vector_type(16))) _Float16 v16h;
typedef __attribute__((ext_vector_type(8)))  _Float16 v8h;
typedef __attribute__((ext_vector_type(8)))  float    v8f;

union U16h { v16h v; v8h h[2]; };

static constexpr int   BATCH   = 8;
static constexpr int   CH      = 256;
static constexpr int   IMG     = 512;
static constexpr int   PRE_NMS = 2000;
static constexpr int   KEEP    = 500;
static constexpr float NMS_THR = 0.2f;
static constexpr float BIGWH   = 120.0f;
static constexpr float MAX_DWH = 4.135166556742356f;  // log(1000/16)

__device__ __forceinline__ unsigned fmono(float f) {
    unsigned u = __float_as_uint(f);
    return (u & 0x80000000u) ? ~u : (u | 0x80000000u);  // monotone ascending map
}

// ---------------------------------------------------------------------------
// Pack conv weights (5,256,256,3,3) f32 -> f16 in WMMA B-fragment order:
// [layer][ko(0..71)][ntile(0..15)][lane(0..31)][half(0..15)]
// B layout (32x16, K x N): lane<16 -> col=lane, K=h ; lane>=16 -> col=lane-16, K=16+h
// ---------------------------------------------------------------------------
__global__ void k_pack_weights(const float* __restrict__ Wb, half_t* __restrict__ wp) {
    int t = blockIdx.x * blockDim.x + threadIdx.x;
    const int total = 5 * 72 * 16 * 32 * 16;
    if (t >= total) return;
    int h     = t & 15;
    int lane  = (t >> 4) & 31;
    int nt    = (t >> 9) & 15;
    int ko    = (t >> 13) % 72;
    int layer = (t >> 13) / 72;
    int cout  = nt * 16 + (lane & 15);
    int koff  = ((lane >> 4) << 4) + h;          // 0..31
    int kg    = ko * 32 + koff;                  // 0..2303
    int tap   = kg >> 8;                         // ky*3+kx
    int cin   = kg & 255;
    float w = Wb[(((size_t)layer * 256 + cout) * 256 + cin) * 9 + tap];
    wp[t] = (half_t)w;
}

// ---------------------------------------------------------------------------
// BN + ReLU + f32->f16, NCHW -> zero-padded NHWC (B, H+2, W+2, 256)
// ---------------------------------------------------------------------------
__global__ void k_bnrelu_pad(const float* __restrict__ x, half_t* __restrict__ y,
                             const float* __restrict__ g, const float* __restrict__ be,
                             const float* __restrict__ mu, const float* __restrict__ var,
                             int H, int W) {
    size_t t = (size_t)blockIdx.x * blockDim.x + threadIdx.x;
    size_t total = (size_t)BATCH * (H + 2) * (W + 2) * CH;
    if (t >= total) return;
    int c = (int)(t & 255);
    size_t r = t >> 8;
    int wp = (int)(r % (W + 2)); r /= (W + 2);
    int hp = (int)(r % (H + 2));
    int b  = (int)(r / (H + 2));
    half_t out = (half_t)0.0f;
    if (hp > 0 && hp <= H && wp > 0 && wp <= W) {
        float xv = x[(((size_t)b * CH + c) * H + (hp - 1)) * W + (wp - 1)];
        float s  = g[c] * rsqrtf(var[c] + 1e-5f);
        float v  = (xv - mu[c]) * s + be[c];
        out = (half_t)fmaxf(v, 0.0f);
    }
    y[t] = out;
}

// ---------------------------------------------------------------------------
// 3x3 conv, implicit GEMM via v_wmma_f32_16x16x32_f16.
// Wave: 16 pixels x 64 couts (4 accumulators, A reused). Block: 8 waves =
// 32 pixels x 256 couts. Per K-chunk the 16 KB weight block is staged into a
// double-buffered LDS tile with global_load_async_to_lds_b128 (ASYNCcnt),
// so B-fragments come from LDS (ds_load_b128) and the vmem pipe serves A only.
// ---------------------------------------------------------------------------
__global__ void __launch_bounds__(256)
k_conv3x3_wmma(const half_t* __restrict__ ypad, const half_t* __restrict__ wpack,
               const float* __restrict__ bias, float* __restrict__ out,
               int H, int W, int layer) {
    __shared__ half_t wsm[2][8192];           // 2 x 16 KB ping-pong weight tiles
    int tid  = threadIdx.x;
    int lane = tid & 31;
    int wave = tid >> 5;
    int pg = wave & 1, cg = wave >> 1;
    int tilesW = W >> 5;
    int wt = blockIdx.x % tilesW;
    int h  = (blockIdx.x / tilesW) % H;
    int b  = blockIdx.x / (tilesW * H);
    int w0 = wt * 32 + pg * 16;
    int lm = lane & 15;
    int hi = lane >> 4;
    const size_t Wp2 = (size_t)(W + 2);

    v8f acc[4] = {v8f{}, v8f{}, v8f{}, v8f{}};

    const half_t* wl = wpack + ((size_t)layer * 72) * 8192;  // 8192 halves per ko

    // Prologue: stage ko=0 weight tile (each thread copies 64 B; offset applies
    // to both global and LDS sides of the async op).
    {
        const half_t* g = wl + (size_t)tid * 32;
        unsigned l = (unsigned)(uintptr_t)&wsm[0][(size_t)tid * 32];
        asm volatile(
            "global_load_async_to_lds_b128 %0, %1, off\n\t"
            "global_load_async_to_lds_b128 %0, %1, off offset:16\n\t"
            "global_load_async_to_lds_b128 %0, %1, off offset:32\n\t"
            "global_load_async_to_lds_b128 %0, %1, off offset:48"
            :: "v"(l), "v"(g) : "memory");
    }

    for (int ko = 0; ko < 72; ++ko) {
        // Current tile's async writes (issued last iteration) are done per-wave,
        // then barrier: all waves' writes visible, all prior readers of the
        // buffer we are about to overwrite have finished.
        asm volatile("s_wait_asynccnt 0" ::: "memory");
        __syncthreads();
        if (ko + 1 < 72) {
            const half_t* g = wl + (size_t)(ko + 1) * 8192 + (size_t)tid * 32;
            unsigned l = (unsigned)(uintptr_t)&wsm[(ko + 1) & 1][(size_t)tid * 32];
            asm volatile(
                "global_load_async_to_lds_b128 %0, %1, off\n\t"
                "global_load_async_to_lds_b128 %0, %1, off offset:16\n\t"
                "global_load_async_to_lds_b128 %0, %1, off offset:32\n\t"
                "global_load_async_to_lds_b128 %0, %1, off offset:48"
                :: "v"(l), "v"(g) : "memory");
        }

        int tap = ko >> 3;
        int dy = tap / 3;
        int dx = tap - dy * 3;
        int cb = (ko & 7) << 5;
        // A fragment: 16x32 f16. lane<16: K=0..15 ; lane>=16: K=16..31
        const half_t* ap = ypad +
            ((((size_t)b * (H + 2) + (h + dy)) * Wp2 + (size_t)(w0 + lm + dx)) << 8)
            + cb + (hi << 4);
        U16h A;
        A.h[0] = *(const v8h*)(ap);
        A.h[1] = *(const v8h*)(ap + 8);
        // B fragments from LDS: 4 ntiles, 32 B contiguous per lane
        const half_t* bp = &wsm[ko & 1][((size_t)cg * 4) * 512 + (size_t)lane * 16];
        U16h B0, B1, B2, B3;
        B0.h[0] = *(const v8h*)(bp +    0); B0.h[1] = *(const v8h*)(bp +    8);
        B1.h[0] = *(const v8h*)(bp +  512); B1.h[1] = *(const v8h*)(bp +  520);
        B2.h[0] = *(const v8h*)(bp + 1024); B2.h[1] = *(const v8h*)(bp + 1032);
        B3.h[0] = *(const v8h*)(bp + 1536); B3.h[1] = *(const v8h*)(bp + 1544);
        acc[0] = __builtin_amdgcn_wmma_f32_16x16x32_f16(false, A.v, false, B0.v, (short)0, acc[0], false, false);
        acc[1] = __builtin_amdgcn_wmma_f32_16x16x32_f16(false, A.v, false, B1.v, (short)0, acc[1], false, false);
        acc[2] = __builtin_amdgcn_wmma_f32_16x16x32_f16(false, A.v, false, B2.v, (short)0, acc[2], false, false);
        acc[3] = __builtin_amdgcn_wmma_f32_16x16x32_f16(false, A.v, false, B3.v, (short)0, acc[3], false, false);
    }
    // D layout: lane holds (M = v + hi*8, N = lane%16); M = pixel offset, N = cout offset
    int coutBase = cg * 64 + lm;
    #pragma unroll
    for (int j = 0; j < 4; ++j) {
        int cout = coutBase + j * 16;
        float bv = bias[cout];
        #pragma unroll
        for (int v = 0; v < 8; ++v) {
            int wpix = w0 + v + hi * 8;
            out[(((size_t)b * H + h) * W + wpix) * CH + cout] = acc[j][v] + bv;
        }
    }
}

// ---------------------------------------------------------------------------
// Heads: wave-per-pixel dot products over 256 channels.
// ---------------------------------------------------------------------------
__global__ void k_head_logits(const float* __restrict__ c, const float* __restrict__ Wlog,
                              const float* __restrict__ blog, float* __restrict__ scores,
                              int HW) {
    int lane = threadIdx.x & 31;
    int pix = blockIdx.x * 8 + (threadIdx.x >> 5);
    if (pix >= BATCH * HW) return;
    const float* cp = c + (size_t)pix * CH + lane * 8;
    float x[8];
    #pragma unroll
    for (int i = 0; i < 8; ++i) x[i] = cp[i];
    float p[3];
    #pragma unroll
    for (int k = 0; k < 3; ++k) {
        const float* wr = Wlog + k * CH + lane * 8;
        float s = 0.f;
        #pragma unroll
        for (int i = 0; i < 8; ++i) s += x[i] * wr[i];
        p[k] = s;
    }
    #pragma unroll
    for (int off = 16; off > 0; off >>= 1) {
        p[0] += __shfl_xor(p[0], off);
        p[1] += __shfl_xor(p[1], off);
        p[2] += __shfl_xor(p[2], off);
    }
    if (lane == 0) {
        int b = pix / HW, hw = pix % HW;
        #pragma unroll
        for (int k = 0; k < 3; ++k) {
            float l = p[k] + blog[k];
            scores[((size_t)b * 3 + k) * HW + hw] = 1.0f / (1.0f + expf(-l));
        }
    }
}

__global__ void k_head_reg(const float* __restrict__ r, const float* __restrict__ Wreg,
                           const float* __restrict__ breg, float* __restrict__ reg,
                           int HW) {
    int lane = threadIdx.x & 31;
    int pix = blockIdx.x * 8 + (threadIdx.x >> 5);
    if (pix >= BATCH * HW) return;
    const float* cp = r + (size_t)pix * CH + lane * 8;
    float x[8];
    #pragma unroll
    for (int i = 0; i < 8; ++i) x[i] = cp[i];
    float p[12];
    #pragma unroll
    for (int k = 0; k < 12; ++k) {
        const float* wr = Wreg + k * CH + lane * 8;
        float s = 0.f;
        #pragma unroll
        for (int i = 0; i < 8; ++i) s += x[i] * wr[i];
        p[k] = s;
    }
    #pragma unroll
    for (int off = 16; off > 0; off >>= 1) {
        #pragma unroll
        for (int k = 0; k < 12; ++k) p[k] += __shfl_xor(p[k], off);
    }
    if (lane == 0) {
        int b = pix / HW, hw = pix % HW;
        #pragma unroll
        for (int k = 0; k < 12; ++k)
            reg[((size_t)b * 12 + k) * HW + hw] = p[k] + breg[k];
    }
}

// ---------------------------------------------------------------------------
// Selection: key = (~mono(masked_score) << 32) | idx  -> ascending sort gives
// descending score, index-ascending ties (matches lax.top_k / stable argsort).
// ---------------------------------------------------------------------------
__global__ void k_build_keys(const float* __restrict__ scores,
                             unsigned long long* __restrict__ keys, int A, int N2) {
    int t = blockIdx.x * blockDim.x + threadIdx.x;
    if (t >= BATCH * N2) return;
    int b = t / N2, n = t % N2;
    unsigned long long key = ~0ull;
    if (n < A) {
        float s = scores[(size_t)b * A + n];
        float m = (s >= 0.5f) ? s : -1.0f;
        key = ((unsigned long long)(~fmono(m)) << 32) | (unsigned)n;
    }
    keys[t] = key;
}

__global__ void k_bitonic_step(unsigned long long* __restrict__ keys, int N2, int k, int j) {
    int t = blockIdx.x * blockDim.x + threadIdx.x;
    if (t >= BATCH * N2) return;
    int b = t / N2, n = t % N2;
    int p = n ^ j;
    if (p <= n) return;
    unsigned long long* kb = keys + (size_t)b * N2;
    unsigned long long a = kb[n], c = kb[p];
    bool asc = ((n & k) == 0);
    if ((a > c) == asc) { kb[n] = c; kb[p] = a; }
}

__global__ void k_proposals(const unsigned long long* __restrict__ keys,
                            const float* __restrict__ scores,
                            const float* __restrict__ reg,
                            const float* __restrict__ anchors,
                            float* __restrict__ boxes4k, float* __restrict__ sc4k,
                            int A, int HW, int N2, int lvl) {
    int t = blockIdx.x * blockDim.x + threadIdx.x;
    if (t >= BATCH * PRE_NMS) return;
    int b = t / PRE_NMS, i = t % PRE_NMS;
    unsigned long long key = keys[(size_t)b * N2 + i];
    unsigned idx = (unsigned)key;
    int slot = b * (2 * PRE_NMS) + lvl * PRE_NMS + i;
    float bx[4] = {0.f, 0.f, 0.f, 0.f};
    float so = -1.0f;
    if (idx < (unsigned)A) {
        float s = scores[(size_t)b * A + idx];
        bool v = (s >= 0.5f);
        int k  = (int)(idx / HW);
        int hw = (int)(idx % HW);
        float d0 = reg[((size_t)b * 12 + 0 + k) * HW + hw];
        float d1 = reg[((size_t)b * 12 + 3 + k) * HW + hw];
        float d2 = reg[((size_t)b * 12 + 6 + k) * HW + hw];
        float d3 = reg[((size_t)b * 12 + 9 + k) * HW + hw];
        const float* ap = anchors + (size_t)idx * 4;
        float a0 = ap[0], a1 = ap[1], a2 = ap[2], a3 = ap[3];
        float wa = a2 - a0, ha = a3 - a1;
        float cx = d0 * wa + a0 + 0.5f * wa;
        float cy = d1 * ha + a1 + 0.5f * ha;
        float w  = expf(fminf(d2, MAX_DWH)) * wa;
        float h  = expf(fminf(d3, MAX_DWH)) * ha;
        float x1 = fminf(fmaxf(cx - 0.5f * w, 0.f), (float)IMG);
        float y1 = fminf(fmaxf(cy - 0.5f * h, 0.f), (float)IMG);
        float x2 = fminf(fmaxf(cx + 0.5f * w, 0.f), (float)IMG);
        float y2 = fminf(fmaxf(cy + 0.5f * h, 0.f), (float)IMG);
        float bw = x2 - x1, bh = y2 - y1;
        v = v && (bw <= BIGWH) && (bh <= BIGWH);
        if (v) { bx[0] = x1; bx[1] = y1; bx[2] = x2; bx[3] = y2; so = s; }
    }
    boxes4k[(size_t)slot * 4 + 0] = bx[0];
    boxes4k[(size_t)slot * 4 + 1] = bx[1];
    boxes4k[(size_t)slot * 4 + 2] = bx[2];
    boxes4k[(size_t)slot * 4 + 3] = bx[3];
    sc4k[slot] = so;
}

// ---------------------------------------------------------------------------
// NMS: one block per batch. LDS bitonic sort of 4096 keys, sequential-greedy
// suppression, stable compaction to 500 zero-padded output slots.
// ---------------------------------------------------------------------------
__global__ void __launch_bounds__(1024)
k_nms(const float* __restrict__ boxes4k, const float* __restrict__ sc4k,
      float* __restrict__ sboxes, float* __restrict__ sscores,
      float* __restrict__ outp) {
    __shared__ unsigned long long skey[4096];
    __shared__ unsigned char     skeep[4096];
    __shared__ int               slots[KEEP];
    __shared__ int               nk;
    const int N = 4000, N2 = 4096;
    int b = blockIdx.x;
    int tid = threadIdx.x;

    for (int n = tid; n < N2; n += 1024) {
        unsigned long long key = ~0ull;
        if (n < N) {
            float s = sc4k[(size_t)b * N + n];
            key = ((unsigned long long)(~fmono(s)) << 32) | (unsigned)n;
        }
        skey[n] = key;
    }
    __syncthreads();
    for (int k = 2; k <= N2; k <<= 1) {
        for (int j = k >> 1; j > 0; j >>= 1) {
            for (int n = tid; n < N2; n += 1024) {
                int p = n ^ j;
                if (p > n) {
                    unsigned long long a = skey[n], c = skey[p];
                    bool asc = ((n & k) == 0);
                    if ((a > c) == asc) { skey[n] = c; skey[p] = a; }
                }
            }
            __syncthreads();
        }
    }
    float* sb = sboxes + (size_t)b * N2 * 4;
    float* ss = sscores + (size_t)b * N2;
    for (int n = tid; n < N2; n += 1024) {
        unsigned idx = (unsigned)skey[n];
        float x1 = 0.f, y1 = 0.f, x2 = 0.f, y2 = 0.f, s = -1.0f;
        if (idx < (unsigned)N) {
            const float* bp = boxes4k + ((size_t)b * N + idx) * 4;
            x1 = bp[0]; y1 = bp[1]; x2 = bp[2]; y2 = bp[3];
            s = sc4k[(size_t)b * N + idx];
        }
        sb[(size_t)n * 4 + 0] = x1; sb[(size_t)n * 4 + 1] = y1;
        sb[(size_t)n * 4 + 2] = x2; sb[(size_t)n * 4 + 3] = y2;
        ss[n] = s;
        skeep[n] = (s >= 0.5f) ? 1 : 0;
    }
    __threadfence();
    __syncthreads();
    // sequential greedy suppression (invalid boxes are all-zero -> IoU 0)
    for (int i = 0; i < N; ++i) {
        if (skeep[i]) {
            float ax1 = sb[(size_t)i * 4], ay1 = sb[(size_t)i * 4 + 1];
            float ax2 = sb[(size_t)i * 4 + 2], ay2 = sb[(size_t)i * 4 + 3];
            float areaA = (ax2 - ax1) * (ay2 - ay1);
            for (int n = i + 1 + tid; n < N; n += 1024) {
                if (skeep[n]) {
                    float bx1 = sb[(size_t)n * 4], by1 = sb[(size_t)n * 4 + 1];
                    float bx2 = sb[(size_t)n * 4 + 2], by2 = sb[(size_t)n * 4 + 3];
                    float areaB = (bx2 - bx1) * (by2 - by1);
                    float iw = fmaxf(0.f, fminf(ax2, bx2) - fmaxf(ax1, bx1));
                    float ih = fmaxf(0.f, fminf(ay2, by2) - fmaxf(ay1, by1));
                    float inter = iw * ih;
                    float iou = inter / (areaA + areaB - inter + 1e-9f);
                    if (iou > NMS_THR) skeep[n] = 0;
                }
            }
        }
        __syncthreads();
    }
    if (tid == 0) {
        int r = 0;
        for (int n = 0; n < N && r < KEEP; ++n)
            if (skeep[n]) slots[r++] = n;
        nk = r;
    }
    __syncthreads();
    float* op = outp + (size_t)b * KEEP * 5;
    for (int s2 = tid; s2 < KEEP; s2 += 1024) {
        float o0 = 0.f, o1 = 0.f, o2 = 0.f, o3 = 0.f, o4 = 0.f;
        if (s2 < nk) {
            int n = slots[s2];
            o0 = sb[(size_t)n * 4];     o1 = sb[(size_t)n * 4 + 1];
            o2 = sb[(size_t)n * 4 + 2]; o3 = sb[(size_t)n * 4 + 3];
            o4 = ss[n];
        }
        op[(size_t)s2 * 5 + 0] = o0; op[(size_t)s2 * 5 + 1] = o1;
        op[(size_t)s2 * 5 + 2] = o2; op[(size_t)s2 * 5 + 3] = o3;
        op[(size_t)s2 * 5 + 4] = o4;
    }
}

// ---------------------------------------------------------------------------
extern "C" void kernel_launch(void* const* d_in, const int* in_sizes, int n_in,
                              void* d_out, int out_size, void* d_ws, size_t ws_size,
                              hipStream_t stream) {
    (void)in_sizes; (void)n_in; (void)out_size; (void)ws_size;
    const float* f4       = (const float*)d_in[0];
    const float* f8       = (const float*)d_in[1];
    const float* anchors0 = (const float*)d_in[2];
    const float* anchors1 = (const float*)d_in[3];
    const float* Wb       = (const float*)d_in[4];
    const float* bb       = (const float*)d_in[5];
    const float* bng      = (const float*)d_in[6];
    const float* bnb      = (const float*)d_in[7];
    const float* bnmu     = (const float*)d_in[8];
    const float* bnvar    = (const float*)d_in[9];
    const float* Wlog     = (const float*)d_in[10];
    const float* blog     = (const float*)d_in[11];
    const float* Wreg     = (const float*)d_in[12];
    const float* breg     = (const float*)d_in[13];
    float* outp = (float*)d_out;

    char* ws = (char*)d_ws;
    auto alloc = [&](size_t bytes) -> char* {
        char* p = ws;
        ws += (bytes + 255) & ~(size_t)255;
        return p;
    };
    half_t* wpack = (half_t*)alloc((size_t)5 * 72 * 16 * 32 * 16 * 2);
    half_t* ypad  = (half_t*)alloc((size_t)BATCH * 130 * 130 * CH * 2);
    float* bufF   = (float*)alloc((size_t)BATCH * 128 * 128 * CH * 4);
    float* bufT   = (float*)alloc((size_t)BATCH * 128 * 128 * CH * 4);
    float* scores = (float*)alloc((size_t)BATCH * 49152 * 4);
    float* regb   = (float*)alloc((size_t)BATCH * 12 * 16384 * 4);
    unsigned long long* keys = (unsigned long long*)alloc((size_t)BATCH * 65536 * 8);
    float* boxes4k = (float*)alloc((size_t)BATCH * 4000 * 4 * 4);
    float* sc4k    = (float*)alloc((size_t)BATCH * 4000 * 4);
    float* sbox    = (float*)alloc((size_t)BATCH * 4096 * 4 * 4);
    float* ssc     = (float*)alloc((size_t)BATCH * 4096 * 4);

    {
        int total = 5 * 72 * 16 * 32 * 16;
        k_pack_weights<<<(total + 255) / 256, 256, 0, stream>>>(Wb, wpack);
    }

    auto run_level = [&](const float* fin, const float* anchors, int H, int lvl) {
        int W = H, HW = H * W, A = 3 * HW;
        int N2 = 1; while (N2 < A) N2 <<= 1;
        auto bnrelu = [&](const float* src, int layer) {
            size_t total = (size_t)BATCH * (H + 2) * (W + 2) * CH;
            k_bnrelu_pad<<<(unsigned)((total + 255) / 256), 256, 0, stream>>>(
                src, ypad, bng + layer * 256, bnb + layer * 256,
                bnmu + layer * 256, bnvar + layer * 256, H, W);
        };
        auto conv = [&](float* dst, int layer) {
            int grid = BATCH * H * (W / 32);
            k_conv3x3_wmma<<<grid, 256, 0, stream>>>(ypad, wpack, bb + layer * 256,
                                                     dst, H, W, layer);
        };
        // f = conv0(bnrelu0(x))
        bnrelu(fin, 0);  conv(bufF, 0);
        // classification branch: c = conv2(bnrelu2(conv1(bnrelu1(f))))
        bnrelu(bufF, 1); conv(bufT, 1);
        bnrelu(bufT, 2); conv(bufT, 2);
        k_head_logits<<<BATCH * HW / 8, 256, 0, stream>>>(bufT, Wlog, blog, scores, HW);
        // regression branch: r = conv4(bnrelu4(conv3(bnrelu3(f))))
        bnrelu(bufF, 3); conv(bufT, 3);
        bnrelu(bufT, 4); conv(bufT, 4);
        k_head_reg<<<BATCH * HW / 8, 256, 0, stream>>>(bufT, Wreg, breg, regb, HW);
        // top-PRE_NMS selection via bitonic sort of packed keys
        {
            int tot = BATCH * N2;
            k_build_keys<<<(tot + 255) / 256, 256, 0, stream>>>(scores, keys, A, N2);
            for (int k = 2; k <= N2; k <<= 1)
                for (int j = k >> 1; j > 0; j >>= 1)
                    k_bitonic_step<<<(tot + 255) / 256, 256, 0, stream>>>(keys, N2, k, j);
        }
        {
            int tot = BATCH * PRE_NMS;
            k_proposals<<<(tot + 255) / 256, 256, 0, stream>>>(
                keys, scores, regb, anchors, boxes4k, sc4k, A, HW, N2, lvl);
        }
    };

    run_level(f4, anchors0, 128, 0);
    run_level(f8, anchors1, 64, 1);

    k_nms<<<BATCH, 1024, 0, stream>>>(boxes4k, sc4k, sbox, ssc, outp);
}
